// AttentionPredictorLSTM_27496380629555
// MI455X (gfx1250) — compile-verified, matrix operand order checked by reading, not run on hardware
//
#include <hip/hip_runtime.h>

// ---------------------------------------------------------------------------
// AttentionPredictorLSTM on MI455X (gfx1250, wave32, WMMA bf16)
// All GEMMs go through v_wmma_f32_16x16x32_bf16. Weights pre-transposed to
// [N][K] bf16 so A and B fragments are two uint4 loads per lane each.
// Each wave register-blocks a 32x64 C tile: 2 A frags reused over 4 B frags
// -> 12 b128 loads per 8 WMMA. K is a template constant and the K-loop fully
// unrolled so all loads use immediate offsets and can hoist across wmmas.
// ---------------------------------------------------------------------------

typedef __bf16 bf16;
typedef __attribute__((ext_vector_type(16))) __bf16 v16bf;
typedef __attribute__((ext_vector_type(8)))  float  v8f;

#define B_SZ   128
#define N1     32
#define V_SZ   31
#define F_SZ   128
#define G4     512
#define NHEAD  8
#define HDIM   16
#define NLANE  64
#define NKEY   96      // 32 agents + 64 lanes
#define HH     20      // history steps (T-2)
#define NPRED  6
#define P6     36
#define P6PAD  64      // padded to N%64==0 for the blocked GEMM

__device__ __forceinline__ float sigmf(float x) { return 1.0f / (1.0f + __expf(-x)); }

// ---------------------------------------------------------------------------
// Blocked bf16 GEMM:  C[M,N] = A[M,K] * Wt[N,K]^T (+bias) (+resid)
// A row-major bf16 (lda), Wt is N-major bf16 (ldw = K stride), C f32 (ldc).
// Block = 4 waves; each wave owns a 32(M) x 64(N) tile: acc[8] v8f.
// Per K-chunk(32): load a0,a1,b0..b3 into distinct regs (12 x b128), then
// 8 wmma. Requires M%32==0, N%64==0, K%32==0.
// Fragment layout (16-bit 16x32): lane<16 -> K0..7,K16..23 ; lane>=16 ->
// K8..15,K24..31 ; row/col = lane&15. C: vgpr i -> row i + 8*(lane>>4).
// ---------------------------------------------------------------------------
template <int K>
__global__ void gemm_bf16_wmma(const bf16* __restrict__ A, int lda,
                               const bf16* __restrict__ Wt, int ldw,
                               float* __restrict__ C, int ldc,
                               const float* __restrict__ bias,
                               const float* __restrict__ resid, int ldr,
                               int M, int N, int Nstore)
{
    const int lane = threadIdx.x;                       // 0..31
    const int m0   = (blockIdx.x * blockDim.y + threadIdx.y) * 32;
    const int n0   = blockIdx.y * 64;
    if (m0 >= M) return;                                // uniform per wave

    const int hf = lane >> 4;                           // 0 or 1
    const int lq = lane & 15;

    const bf16* arow0 = A + (size_t)(m0 + lq) * lda + hf * 8;
    const bf16* arow1 = arow0 + (size_t)16 * lda;
    const bf16* brow0 = Wt + (size_t)(n0 + lq) * ldw + hf * 8;
    const bf16* brow1 = brow0 + (size_t)16 * ldw;
    const bf16* brow2 = brow0 + (size_t)32 * ldw;
    const bf16* brow3 = brow0 + (size_t)48 * ldw;

    v8f acc[8] = {};
    union F { uint4 u[2]; v16bf v; };
    #pragma unroll
    for (int kk = 0; kk < K; kk += 32) {
        F a0, a1, b0, b1, b2, b3;
        a0.u[0] = *(const uint4*)(arow0 + kk);
        a0.u[1] = *(const uint4*)(arow0 + kk + 16);
        a1.u[0] = *(const uint4*)(arow1 + kk);
        a1.u[1] = *(const uint4*)(arow1 + kk + 16);
        b0.u[0] = *(const uint4*)(brow0 + kk);
        b0.u[1] = *(const uint4*)(brow0 + kk + 16);
        b1.u[0] = *(const uint4*)(brow1 + kk);
        b1.u[1] = *(const uint4*)(brow1 + kk + 16);
        b2.u[0] = *(const uint4*)(brow2 + kk);
        b2.u[1] = *(const uint4*)(brow2 + kk + 16);
        b3.u[0] = *(const uint4*)(brow3 + kk);
        b3.u[1] = *(const uint4*)(brow3 + kk + 16);
        acc[0] = __builtin_amdgcn_wmma_f32_16x16x32_bf16(false, a0.v, false, b0.v, (short)0, acc[0], false, false);
        acc[4] = __builtin_amdgcn_wmma_f32_16x16x32_bf16(false, a1.v, false, b0.v, (short)0, acc[4], false, false);
        acc[1] = __builtin_amdgcn_wmma_f32_16x16x32_bf16(false, a0.v, false, b1.v, (short)0, acc[1], false, false);
        acc[5] = __builtin_amdgcn_wmma_f32_16x16x32_bf16(false, a1.v, false, b1.v, (short)0, acc[5], false, false);
        acc[2] = __builtin_amdgcn_wmma_f32_16x16x32_bf16(false, a0.v, false, b2.v, (short)0, acc[2], false, false);
        acc[6] = __builtin_amdgcn_wmma_f32_16x16x32_bf16(false, a1.v, false, b2.v, (short)0, acc[6], false, false);
        acc[3] = __builtin_amdgcn_wmma_f32_16x16x32_bf16(false, a0.v, false, b3.v, (short)0, acc[3], false, false);
        acc[7] = __builtin_amdgcn_wmma_f32_16x16x32_bf16(false, a1.v, false, b3.v, (short)0, acc[7], false, false);
    }

    #pragma unroll
    for (int s = 0; s < 4; ++s) {
        const int col = n0 + s * 16 + lq;
        if (col < Nstore) {
            const float bv = bias ? bias[col] : 0.0f;
            #pragma unroll
            for (int i = 0; i < 8; ++i) {
                const int row0 = m0 + i + hf * 8;
                float v0 = acc[s][i] + bv;
                float v1 = acc[4 + s][i] + bv;
                if (resid) {
                    v0 += resid[(size_t)row0 * ldr + col];
                    v1 += resid[(size_t)(row0 + 16) * ldr + col];
                }
                C[(size_t)row0 * ldc + col]        = v0;
                C[(size_t)(row0 + 16) * ldc + col] = v1;
            }
        }
    }
}

// ---------------------------------------------------------------------------
// Conv feature extractor: feat[t,b,n,f] = sum_{c,k} input[t+k,b,n,c]*w[f,c,k]+b
// ---------------------------------------------------------------------------
__global__ void conv_feat_kernel(const float* __restrict__ in,
                                 const float* __restrict__ w,
                                 const float* __restrict__ bias,
                                 float* __restrict__ feat)
{
    int idx = blockIdx.x * 256 + threadIdx.x;
    if (idx >= HH * B_SZ * N1 * F_SZ) return;
    int f = idx & 127; int r = idx >> 7;
    int n = r & 31; r >>= 5;
    int b = r & 127; int t = r >> 7;
    float acc = bias[f];
    #pragma unroll
    for (int k = 0; k < 3; ++k) {
        const float* x = in + (((size_t)(t + k) * B_SZ + b) * N1 + n) * 2;
        acc += x[0] * w[f * 6 + k] + x[1] * w[f * 6 + 3 + k];
    }
    feat[idx] = acc;
}

// ---------------------------------------------------------------------------
// Fused lane encoder: relu(mlp1) -> relu(mlp2) -> mean over 10 pts -> bf16
// One block per (b, lane), 64 threads.
// ---------------------------------------------------------------------------
__global__ void lane_encode_kernel(const float* __restrict__ lane_in,
                                   const float* __restrict__ w1, const float* __restrict__ b1,
                                   const float* __restrict__ w2, const float* __restrict__ b2,
                                   bf16* __restrict__ lanes_bf)
{
    int b = blockIdx.x, l = blockIdx.y;
    int tid = threadIdx.x;                        // 0..63
    __shared__ float h1[10 * 64];
    const float* li = lane_in + ((size_t)(b * NLANE + l)) * 10 * 2;
    for (int p = 0; p < 10; ++p) {
        float v = li[p * 2] * w1[tid] + li[p * 2 + 1] * w1[64 + tid] + b1[tid];
        h1[p * 64 + tid] = fmaxf(v, 0.0f);
    }
    __syncthreads();
    float acc = 0.0f;
    for (int p = 0; p < 10; ++p) {
        float v = b2[tid];
        for (int k = 0; k < 64; ++k) v += h1[p * 64 + k] * w2[k * 64 + tid];
        acc += fmaxf(v, 0.0f);
    }
    lanes_bf[((size_t)(b * NLANE + l)) * 64 + tid] = (bf16)(acc * 0.1f);
}

// ---------------------------------------------------------------------------
// Weight prep
// ---------------------------------------------------------------------------
__global__ void wt_transpose(const float* __restrict__ src, bf16* __restrict__ dst,
                             int K, int N)           // src [K][N] -> dst [N][K]
{
    int idx = blockIdx.x * 256 + threadIdx.x;
    if (idx >= N * K) return;
    int n = idx / K, k = idx % K;
    dst[idx] = (bf16)src[(size_t)k * N + n];
}

__global__ void wt_transpose_pad(const float* __restrict__ src, bf16* __restrict__ dst,
                                 int K, int N, int Npad)
{
    int idx = blockIdx.x * 256 + threadIdx.x;
    if (idx >= Npad * K) return;
    int n = idx / K, k = idx % K;
    dst[idx] = (bf16)((n < N) ? src[(size_t)k * N + n] : 0.0f);
}

// LSTM: Wt[512][256] = [wih | whh] rows (already N-major in source), bias = bih+bhh
__global__ void pack_lstm_w(const float* __restrict__ wih, const float* __restrict__ whh,
                            const float* __restrict__ bih, const float* __restrict__ bhh,
                            bf16* __restrict__ Wt, float* __restrict__ bias)
{
    int idx = blockIdx.x * 256 + threadIdx.x;
    if (idx >= G4 * 256) return;
    int n = idx >> 8, kk = idx & 255;
    Wt[idx] = (bf16)((kk < 128) ? wih[(size_t)n * 128 + kk] : whh[(size_t)n * 128 + kk - 128]);
    if (kk == 0) bias[n] = bih[n] + bhh[n];
}

__global__ void f32_to_bf16_kernel(const float* __restrict__ s, bf16* __restrict__ d, int n)
{
    int idx = blockIdx.x * 256 + threadIdx.x;
    if (idx < n) d[idx] = (bf16)s[idx];
}

// ---------------------------------------------------------------------------
// State init
// ---------------------------------------------------------------------------
__global__ void zero_f32(float* p, int n)
{
    int idx = blockIdx.x * 256 + threadIdx.x;
    if (idx < n) p[idx] = 0.0f;
}
__global__ void zero_hpack(bf16* A, int rows)     // zero h slot (cols 128..255)
{
    int idx = blockIdx.x * 256 + threadIdx.x;
    if (idx >= rows * 128) return;
    A[(size_t)(idx >> 7) * 256 + 128 + (idx & 127)] = (bf16)0.0f;
}
__global__ void init_positions(const float* __restrict__ ip, float* pe, float* pv)
{
    int idx = blockIdx.x * 256 + threadIdx.x;
    if (idx >= B_SZ * N1) return;
    int b = idx >> 5, n = idx & 31;
    const float* s = ip + ((size_t)b * N1 + n) * 2;
    if (n == 0) { pe[b * 2] = s[0]; pe[b * 2 + 1] = s[1]; }
    else { float* d = pv + ((size_t)(b * V_SZ + n - 1)) * 2; d[0] = s[0]; d[1] = s[1]; }
}
__global__ void expand_pos(const float* pe, const float* pv, float* pe6, float* pv6)
{
    int idx = blockIdx.x * 256 + threadIdx.x;
    if (idx >= B_SZ * N1 * NPRED) return;
    int p = idx % NPRED; int r = idx / NPRED;
    int b = r >> 5, n = r & 31;
    if (n == 0) {
        pe6[((size_t)b * NPRED + p) * 2]     = pe[b * 2];
        pe6[((size_t)b * NPRED + p) * 2 + 1] = pe[b * 2 + 1];
    } else {
        size_t rr = (size_t)(b * V_SZ + n - 1);
        pv6[(rr * NPRED + p) * 2]     = pv[rr * 2];
        pv6[(rr * NPRED + p) * 2 + 1] = pv[rr * 2 + 1];
    }
}

// ---------------------------------------------------------------------------
// History position integration: pos += (dl*cos(yaw), dl*sin(yaw))
// ---------------------------------------------------------------------------
__global__ void pos_update_hist(const float* __restrict__ in, int t, float* pe, float* pv)
{
    int idx = blockIdx.x * 256 + threadIdx.x;
    if (idx >= B_SZ * N1) return;
    int b = idx >> 5, n = idx & 31;
    const float* x = in + (((size_t)t * B_SZ + b) * N1 + n) * 2;
    float dl = x[0], yaw = x[1];
    float* p = (n == 0) ? pe + b * 2 : pv + ((size_t)(b * V_SZ + n - 1)) * 2;
    p[0] += dl * __cosf(yaw);
    p[1] += dl * __sinf(yaw);
}

// ---------------------------------------------------------------------------
// Token build: tok = base + pos @ pos_w + pos_b ; pos averaged over nMean
// hist: base = feat slice (b,n,f); pred: base = hx_ego / hx_veh
// ---------------------------------------------------------------------------
__global__ void build_tok(const float* __restrict__ featSlice,
                          const float* __restrict__ he, const float* __restrict__ hv,
                          const float* __restrict__ pe, const float* __restrict__ pv, int nMean,
                          const float* __restrict__ pos_w, const float* __restrict__ pos_b,
                          float* __restrict__ tok, bf16* __restrict__ tok_bf)
{
    int idx = blockIdx.x * 256 + threadIdx.x;
    if (idx >= B_SZ * N1 * F_SZ) return;
    int f = idx & 127; int row = idx >> 7;
    int b = row >> 5, n = row & 31;
    float base;
    if (featSlice) base = featSlice[(size_t)row * F_SZ + f];
    else base = (n == 0) ? he[(size_t)b * F_SZ + f]
                         : hv[((size_t)(b * V_SZ + n - 1)) * F_SZ + f];
    const float* p = (n == 0) ? pe + (size_t)b * 2 * nMean
                              : pv + (size_t)(b * V_SZ + n - 1) * 2 * nMean;
    float px = 0.0f, py = 0.0f;
    for (int q = 0; q < nMean; ++q) { px += p[q * 2]; py += p[q * 2 + 1]; }
    px /= (float)nMean; py /= (float)nMean;
    float t = base + px * pos_w[f] + py * pos_w[F_SZ + f] + pos_b[f];
    tok[idx] = t;
    tok_bf[idx] = (bf16)t;
}

// ---------------------------------------------------------------------------
// Fused attention core per (b, head): scores (K=16) + mask + softmax + A@V.
// Agent K/V from fused qkv buffer, lane K/V precomputed (time invariant).
// ---------------------------------------------------------------------------
__global__ void attention_kernel(const float* __restrict__ qkv,
                                 const float* __restrict__ k_lane,
                                 const float* __restrict__ v_lane,
                                 const float* __restrict__ mask_in,
                                 const float* __restrict__ lane_mask,
                                 bf16* __restrict__ attn_o)
{
    const int b = blockIdx.x, h = blockIdx.y;
    const int tid = threadIdx.x;                       // 128 threads
    __shared__ float sQ[N1 * HDIM];
    __shared__ float sK[NKEY * HDIM];
    __shared__ float sV[NKEY * HDIM];
    __shared__ float sS[N1 * NKEY];

    for (int idx = tid; idx < N1 * HDIM; idx += 128) {
        int n = idx >> 4, d = idx & 15;
        sQ[idx] = qkv[((size_t)(b * N1 + n)) * 384 + h * HDIM + d];
    }
    for (int idx = tid; idx < NKEY * HDIM; idx += 128) {
        int n = idx >> 4, d = idx & 15;
        float kv, vv;
        if (n < N1) {
            size_t base = ((size_t)(b * N1 + n)) * 384 + h * HDIM + d;
            kv = qkv[base + 128];
            vv = qkv[base + 256];
        } else {
            size_t base = ((size_t)(b * NLANE + n - N1)) * F_SZ + h * HDIM + d;
            kv = k_lane[base];
            vv = v_lane[base];
        }
        sK[idx] = kv; sV[idx] = vv;
    }
    __syncthreads();

    for (int idx = tid; idx < N1 * NKEY; idx += 128) {
        int qi = idx / NKEY, kj = idx % NKEY;
        float m = (kj < N1) ? mask_in[b * N1 + kj] : lane_mask[b * NLANE + kj - N1];
        float s = -1.0e9f;
        if (m > 0.0f) {
            s = 0.0f;
            #pragma unroll
            for (int d = 0; d < HDIM; ++d) s += sQ[qi * HDIM + d] * sK[kj * HDIM + d];
            s *= 0.25f;                                 // 1/sqrt(16)
        }
        sS[idx] = s;
    }
    __syncthreads();

    if (tid < N1) {
        float mx = -1.0e30f;
        for (int k = 0; k < NKEY; ++k) mx = fmaxf(mx, sS[tid * NKEY + k]);
        float sum = 0.0f;
        for (int k = 0; k < NKEY; ++k) {
            float e = __expf(sS[tid * NKEY + k] - mx);
            sS[tid * NKEY + k] = e; sum += e;
        }
        float inv = 1.0f / sum;
        for (int k = 0; k < NKEY; ++k) sS[tid * NKEY + k] *= inv;
    }
    __syncthreads();

    for (int idx = tid; idx < N1 * HDIM; idx += 128) {
        int qi = idx >> 4, d = idx & 15;
        float o = 0.0f;
        for (int k = 0; k < NKEY; ++k) o += sS[qi * NKEY + k] * sV[k * HDIM + d];
        attn_o[((size_t)(b * N1 + qi)) * F_SZ + h * HDIM + d] = (bf16)o;
    }
}

// ---------------------------------------------------------------------------
// LayerNorm over 128 features; output written (bf16) into packed LSTM input.
// ---------------------------------------------------------------------------
__global__ void layernorm_pack(const float* __restrict__ tokr,
                               const float* __restrict__ g_ego, const float* __restrict__ b_ego,
                               const float* __restrict__ g_veh, const float* __restrict__ b_veh,
                               bf16* __restrict__ A_ego, bf16* __restrict__ A_veh)
{
    const int row = blockIdx.x;                        // b*32+n
    const int b = row >> 5, n = row & 31;
    const int tid = threadIdx.x;                       // 128
    float x = tokr[(size_t)row * F_SZ + tid];
    __shared__ float red[128];
    red[tid] = x; __syncthreads();
    for (int s = 64; s > 0; s >>= 1) { if (tid < s) red[tid] += red[tid + s]; __syncthreads(); }
    float mean = red[0] * (1.0f / 128.0f);
    __syncthreads();
    float d = x - mean;
    red[tid] = d * d; __syncthreads();
    for (int s = 64; s > 0; s >>= 1) { if (tid < s) red[tid] += red[tid + s]; __syncthreads(); }
    float var = red[0] * (1.0f / 128.0f);
    const float* g  = (n == 0) ? g_ego : g_veh;
    const float* bb = (n == 0) ? b_ego : b_veh;
    float y = d * rsqrtf(var + 1e-5f) * g[tid] + bb[tid];
    bf16* dst = (n == 0) ? (A_ego + (size_t)b * 256 + tid)
                         : (A_veh + ((size_t)(b * V_SZ + n - 1)) * 256 + tid);
    *dst = (bf16)y;
}

// ---------------------------------------------------------------------------
// LSTM pointwise: c,h update; h written f32 (for tokens/out) + bf16 into pack.
// ---------------------------------------------------------------------------
__global__ void lstm_pointwise(const float* __restrict__ gates,
                               float* __restrict__ hx, float* __restrict__ cx,
                               bf16* __restrict__ Apack, int rows)
{
    int idx = blockIdx.x * 256 + threadIdx.x;
    if (idx >= rows * F_SZ) return;
    int r = idx >> 7, j = idx & 127;
    const float* g = gates + (size_t)r * G4;
    float gi = sigmf(g[j]);
    float gf = sigmf(g[128 + j]);
    float gg = tanhf(g[256 + j]);
    float go = sigmf(g[384 + j]);
    float c = gf * cx[idx] + gi * gg;
    float h = go * tanhf(c);
    cx[idx] = c; hx[idx] = h;
    Apack[(size_t)r * 256 + 128 + j] = (bf16)h;
}

// ---------------------------------------------------------------------------
// Prediction update: integrate (dl,yaw), emit y = [x,y,o2..o5]
// ---------------------------------------------------------------------------
__global__ void pred_update(const float* __restrict__ oe, const float* __restrict__ ov,
                            float* __restrict__ pe6, float* __restrict__ pv6,
                            float* __restrict__ out, int step)
{
    int idx = blockIdx.x * 256 + threadIdx.x;
    if (idx >= B_SZ * N1 * NPRED) return;
    int p = idx % NPRED; int r = idx / NPRED;
    int b = r >> 5, n = r & 31;
    const float* o = (n == 0) ? oe + (size_t)b * P6 + p * 6
                              : ov + ((size_t)(b * V_SZ + n - 1)) * P6 + p * 6;
    float dl = o[0], yaw = o[1];
    float* pp = (n == 0) ? pe6 + ((size_t)b * NPRED + p) * 2
                         : pv6 + (((size_t)(b * V_SZ + n - 1)) * NPRED + p) * 2;
    float x = pp[0] + dl * __cosf(yaw);
    float y = pp[1] + dl * __sinf(yaw);
    pp[0] = x; pp[1] = y;
    float* yo = out + ((((size_t)step * B_SZ + b) * N1 + n) * NPRED + p) * 6;
    yo[0] = x; yo[1] = y; yo[2] = o[2]; yo[3] = o[3]; yo[4] = o[4]; yo[5] = o[5];
}

// ---------------------------------------------------------------------------
// Host orchestration
// ---------------------------------------------------------------------------
extern "C" void kernel_launch(void* const* d_in, const int* in_sizes, int n_in,
                              void* d_out, int out_size, void* d_ws, size_t ws_size,
                              hipStream_t stream)
{
    const float* in_input    = (const float*)d_in[0];
    const float* in_init_pos = (const float*)d_in[1];
    const float* in_lane_in  = (const float*)d_in[2];
    const float* in_mask     = (const float*)d_in[3];
    const float* in_lmask    = (const float*)d_in[4];
    const float* in_conv_w   = (const float*)d_in[5];
    const float* in_conv_b   = (const float*)d_in[6];
    const float* in_pos_w    = (const float*)d_in[7];
    const float* in_pos_b    = (const float*)d_in[8];
    const float* in_lane_w1  = (const float*)d_in[9];
    const float* in_lane_b1  = (const float*)d_in[10];
    const float* in_lane_w2  = (const float*)d_in[11];
    const float* in_lane_b2  = (const float*)d_in[12];
    const float* in_lane2f   = (const float*)d_in[13];
    const float* in_wq       = (const float*)d_in[14];
    const float* in_wk       = (const float*)d_in[15];
    const float* in_wv       = (const float*)d_in[16];
    const float* in_wo       = (const float*)d_in[17];
    const float* in_lng_e    = (const float*)d_in[18];
    const float* in_lnb_e    = (const float*)d_in[19];
    const float* in_lng      = (const float*)d_in[20];
    const float* in_lnb      = (const float*)d_in[21];
    const float* in_ego_wih  = (const float*)d_in[22];
    const float* in_ego_whh  = (const float*)d_in[23];
    const float* in_ego_bih  = (const float*)d_in[24];
    const float* in_ego_bhh  = (const float*)d_in[25];
    const float* in_veh_wih  = (const float*)d_in[26];
    const float* in_veh_whh  = (const float*)d_in[27];
    const float* in_veh_bih  = (const float*)d_in[28];
    const float* in_veh_bhh  = (const float*)d_in[29];
    const float* in_oe_w     = (const float*)d_in[30];
    const float* in_oe_b     = (const float*)d_in[31];
    const float* in_ov_w     = (const float*)d_in[32];
    const float* in_ov_b     = (const float*)d_in[33];
    (void)in_sizes; (void)n_in; (void)ws_size;

    float* out = (float*)d_out;
    // len_pred is a device scalar (unreadable under graph capture); derive it
    // from out_size: out is (len, B, 32, 6, 6).
    const int len_pred = out_size / (B_SZ * N1 * NPRED * 6);

    // ---- workspace layout ----
    char* ws = (char*)d_ws;
    size_t cur = 0;
    auto alloc = [&](size_t bytes) -> char* {
        char* p = ws + cur;
        cur = (cur + bytes + 255) & ~(size_t)255;
        return p;
    };
    const int MTOK = B_SZ * N1;     // 4096
    const int MVEH = B_SZ * V_SZ;   // 3968
    const int MLAN = B_SZ * NLANE;  // 8192

    float* feat      = (float*)alloc((size_t)HH * MTOK * F_SZ * 4);
    float* tok       = (float*)alloc((size_t)MTOK * F_SZ * 4);
    bf16*  tok_bf    = (bf16*) alloc((size_t)MTOK * F_SZ * 2);
    float* qkv       = (float*)alloc((size_t)MTOK * 384 * 4);
    bf16*  attn_o    = (bf16*) alloc((size_t)MTOK * F_SZ * 2);
    float* tokr      = (float*)alloc((size_t)MTOK * F_SZ * 4);
    bf16*  A_ego     = (bf16*) alloc((size_t)B_SZ * 256 * 2);
    bf16*  A_veh     = (bf16*) alloc((size_t)MVEH * 256 * 2);
    float* gates_e   = (float*)alloc((size_t)B_SZ * G4 * 4);
    float* gates_v   = (float*)alloc((size_t)MVEH * G4 * 4);
    float* hx_ego    = (float*)alloc((size_t)B_SZ * F_SZ * 4);
    float* cx_ego    = (float*)alloc((size_t)B_SZ * F_SZ * 4);
    float* hx_veh    = (float*)alloc((size_t)MVEH * F_SZ * 4);
    float* cx_veh    = (float*)alloc((size_t)MVEH * F_SZ * 4);
    float* pe_h      = (float*)alloc((size_t)B_SZ * 2 * 4);
    float* pv_h      = (float*)alloc((size_t)MVEH * 2 * 4);
    float* pe6       = (float*)alloc((size_t)B_SZ * NPRED * 2 * 4);
    float* pv6       = (float*)alloc((size_t)MVEH * NPRED * 2 * 4);
    bf16*  lanes_bf  = (bf16*) alloc((size_t)MLAN * 64 * 2);
    float* lane_tok  = (float*)alloc((size_t)MLAN * F_SZ * 4);
    bf16*  lane_tbf  = (bf16*) alloc((size_t)MLAN * F_SZ * 2);
    float* k_lane    = (float*)alloc((size_t)MLAN * F_SZ * 4);
    float* v_lane    = (float*)alloc((size_t)MLAN * F_SZ * 4);
    float* oe        = (float*)alloc((size_t)B_SZ * P6 * 4);
    float* ov        = (float*)alloc((size_t)MVEH * P6 * 4);
    bf16*  Wqkv_t    = (bf16*) alloc((size_t)384 * 128 * 2);
    bf16*  wo_t      = (bf16*) alloc((size_t)128 * 128 * 2);
    bf16*  Wego_t    = (bf16*) alloc((size_t)G4 * 256 * 2);
    bf16*  Wveh_t    = (bf16*) alloc((size_t)G4 * 256 * 2);
    float* bias_e    = (float*)alloc((size_t)G4 * 4);
    float* bias_v    = (float*)alloc((size_t)G4 * 4);
    bf16*  Woe_t     = (bf16*) alloc((size_t)P6PAD * 128 * 2);
    bf16*  Wov_t     = (bf16*) alloc((size_t)P6PAD * 128 * 2);
    bf16*  L2F_t     = (bf16*) alloc((size_t)128 * 64 * 2);

    auto gemm = [&](const bf16* A, int lda, const bf16* Wt, int ldw,
                    float* C, int ldc, const float* bias,
                    const float* resid, int ldr, int M, int N, int K, int Nstore) {
        // wave tile 32x64; block = 4 waves stacked along M
        dim3 g((M / 32 + 3) / 4, N / 64), blk(32, 4);
        if (K == 64)
            gemm_bf16_wmma<64><<<g, blk, 0, stream>>>(A, lda, Wt, ldw, C, ldc, bias,
                                                      resid, ldr, M, N, Nstore);
        else if (K == 128)
            gemm_bf16_wmma<128><<<g, blk, 0, stream>>>(A, lda, Wt, ldw, C, ldc, bias,
                                                       resid, ldr, M, N, Nstore);
        else
            gemm_bf16_wmma<256><<<g, blk, 0, stream>>>(A, lda, Wt, ldw, C, ldc, bias,
                                                       resid, ldr, M, N, Nstore);
    };
    auto g1 = [](int n) { return dim3((n + 255) / 256); };

    // ================= prep (once per launch) =================
    wt_transpose<<<g1(128 * 128), 256, 0, stream>>>(in_wq, Wqkv_t,            128, 128);
    wt_transpose<<<g1(128 * 128), 256, 0, stream>>>(in_wk, Wqkv_t + 128 * 128, 128, 128);
    wt_transpose<<<g1(128 * 128), 256, 0, stream>>>(in_wv, Wqkv_t + 256 * 128, 128, 128);
    wt_transpose<<<g1(128 * 128), 256, 0, stream>>>(in_wo, wo_t,               128, 128);
    wt_transpose<<<g1(128 * 64),  256, 0, stream>>>(in_lane2f, L2F_t,          64, 128);
    wt_transpose_pad<<<g1(P6PAD * 128), 256, 0, stream>>>(in_oe_w, Woe_t, 128, P6, P6PAD);
    wt_transpose_pad<<<g1(P6PAD * 128), 256, 0, stream>>>(in_ov_w, Wov_t, 128, P6, P6PAD);
    pack_lstm_w<<<g1(G4 * 256), 256, 0, stream>>>(in_ego_wih, in_ego_whh, in_ego_bih, in_ego_bhh, Wego_t, bias_e);
    pack_lstm_w<<<g1(G4 * 256), 256, 0, stream>>>(in_veh_wih, in_veh_whh, in_veh_bih, in_veh_bhh, Wveh_t, bias_v);

    conv_feat_kernel<<<g1(HH * MTOK * F_SZ), 256, 0, stream>>>(in_input, in_conv_w, in_conv_b, feat);

    lane_encode_kernel<<<dim3(B_SZ, NLANE), 64, 0, stream>>>(in_lane_in, in_lane_w1, in_lane_b1,
                                                             in_lane_w2, in_lane_b2, lanes_bf);
    gemm(lanes_bf, 64, L2F_t, 64, lane_tok, 128, nullptr, nullptr, 0, MLAN, 128, 64, 128);
    f32_to_bf16_kernel<<<g1(MLAN * F_SZ), 256, 0, stream>>>(lane_tok, lane_tbf, MLAN * F_SZ);
    // time-invariant lane K/V (hoisted out of the 50-step recurrence)
    gemm(lane_tbf, 128, Wqkv_t + 128 * 128, 128, k_lane, 128, nullptr, nullptr, 0, MLAN, 128, 128, 128);
    gemm(lane_tbf, 128, Wqkv_t + 256 * 128, 128, v_lane, 128, nullptr, nullptr, 0, MLAN, 128, 128, 128);

    zero_f32<<<g1(B_SZ * F_SZ), 256, 0, stream>>>(hx_ego, B_SZ * F_SZ);
    zero_f32<<<g1(B_SZ * F_SZ), 256, 0, stream>>>(cx_ego, B_SZ * F_SZ);
    zero_f32<<<g1(MVEH * F_SZ), 256, 0, stream>>>(hx_veh, MVEH * F_SZ);
    zero_f32<<<g1(MVEH * F_SZ), 256, 0, stream>>>(cx_veh, MVEH * F_SZ);
    zero_hpack<<<g1(B_SZ * 128), 256, 0, stream>>>(A_ego, B_SZ);
    zero_hpack<<<g1(MVEH * 128), 256, 0, stream>>>(A_veh, MVEH);
    init_positions<<<g1(B_SZ * N1), 256, 0, stream>>>(in_init_pos, pe_h, pv_h);

    // one recurrent step (shared by history and prediction)
    auto cell_step = [&]() {
        gemm(tok_bf, 128, Wqkv_t, 128, qkv, 384, nullptr, nullptr, 0, MTOK, 384, 128, 384);
        attention_kernel<<<dim3(B_SZ, NHEAD), 128, 0, stream>>>(qkv, k_lane, v_lane,
                                                                in_mask, in_lmask, attn_o);
        gemm(attn_o, 128, wo_t, 128, tokr, 128, nullptr, tok, 128, MTOK, 128, 128, 128);
        layernorm_pack<<<MTOK, 128, 0, stream>>>(tokr, in_lng_e, in_lnb_e, in_lng, in_lnb, A_ego, A_veh);
        gemm(A_ego, 256, Wego_t, 256, gates_e, G4, bias_e, nullptr, 0, B_SZ, G4, 256, G4);
        gemm(A_veh, 256, Wveh_t, 256, gates_v, G4, bias_v, nullptr, 0, MVEH, G4, 256, G4);
        lstm_pointwise<<<g1(B_SZ * F_SZ), 256, 0, stream>>>(gates_e, hx_ego, cx_ego, A_ego, B_SZ);
        lstm_pointwise<<<g1(MVEH * F_SZ), 256, 0, stream>>>(gates_v, hx_veh, cx_veh, A_veh, MVEH);
    };

    // ================= history scan (20 steps) =================
    for (int t = 0; t < HH; ++t) {
        pos_update_hist<<<g1(B_SZ * N1), 256, 0, stream>>>(in_input, t, pe_h, pv_h);
        build_tok<<<g1(MTOK * F_SZ), 256, 0, stream>>>(feat + (size_t)t * MTOK * F_SZ,
                                                       nullptr, nullptr, pe_h, pv_h, 1,
                                                       in_pos_w, in_pos_b, tok, tok_bf);
        cell_step();
    }

    expand_pos<<<g1(B_SZ * N1 * NPRED), 256, 0, stream>>>(pe_h, pv_h, pe6, pv6);

    // ================= prediction scan (len_pred steps) =================
    for (int p = 0; p < len_pred; ++p) {
        build_tok<<<g1(MTOK * F_SZ), 256, 0, stream>>>(nullptr, hx_ego, hx_veh,
                                                       pe6, pv6, NPRED,
                                                       in_pos_w, in_pos_b, tok, tok_bf);
        cell_step();
        // out projections read packed bf16 h in place (lda=256)
        gemm(A_ego + 128, 256, Woe_t, 128, oe, P6, in_oe_b, nullptr, 0, B_SZ, P6PAD, 128, P6);
        gemm(A_veh + 128, 256, Wov_t, 128, ov, P6, in_ov_b, nullptr, 0, MVEH, P6PAD, 128, P6);
        pred_update<<<g1(B_SZ * N1 * NPRED), 256, 0, stream>>>(oe, ov, pe6, pv6, out, p);
    }
}
